// SpiralAttention_74577812127891
// MI455X (gfx1250) — compile-verified, matrix operand order checked by reading, not run on hardware
//
#include <hip/hip_runtime.h>
#include <hip/hip_bf16.h>

#define C_DIM 1024
#define T_DIM 1024
#define H_DIM 16
#define D_DIM 64
#define B_DIM 4
#define M_ROWS (B_DIM * T_DIM) /* 4096 */

typedef __attribute__((ext_vector_type(16))) _Float16 v16h;
typedef __attribute__((ext_vector_type(8)))  _Float16 v8h;
typedef __attribute__((ext_vector_type(8)))  float    v8f;
typedef __attribute__((ext_vector_type(4)))  int      v4i;

typedef __attribute__((address_space(1))) v4i* gas_v4i;
typedef __attribute__((address_space(3))) v4i* las_v4i;

#if defined(__has_builtin)
#if __has_builtin(__builtin_amdgcn_global_load_async_to_lds_b128)
#define HAVE_ASYNC_LDS 1
#endif
#endif

#define WMMA_F16(A_, B_, C_) \
  __builtin_amdgcn_wmma_f32_16x16x32_f16(false, (A_), false, (B_), (short)0, (C_), false, false)

// Load one 16x32 f16 A/B fragment (this lane's share): per CDNA5 ISA layout,
// lanes 0-15 hold K 0..7 and 16..23, lanes 16-31 hold K 8..15 and 24..31.
// Caller folds (lane&15)*ld + (lane>>4)*8 into p. Works for global or LDS.
__device__ __forceinline__ v16h frag_ld(const _Float16* p) {
  v8h lo = *(const v8h*)(p);
  v8h hi = *(const v8h*)(p + 16);
  v16h r;
#pragma unroll
  for (int i = 0; i < 8; ++i) { r[i] = lo[i]; r[i + 8] = hi[i]; }
  return r;
}

// Stage a 16-row x 1024-col f16 A tile (32 KB) into LDS, shared by the block.
// Uses the gfx1250 async global->LDS DMA path when the builtin is available.
__device__ __forceinline__ void stage_a_tile(const _Float16* __restrict__ src,
                                             _Float16* lds_tile) {
  const int tid = threadIdx.x; // 128 threads, 16 chunks of 16B each
#pragma unroll
  for (int c = 0; c < 16; ++c) {
    const int o = (c * 128 + tid) * 8; // half-element offset
#if defined(HAVE_ASYNC_LDS)
    __builtin_amdgcn_global_load_async_to_lds_b128(
        (gas_v4i)(src + o), (las_v4i)(lds_tile + o), 0, 0);
#else
    *(v8h*)(lds_tile + o) = *(const v8h*)(src + o);
#endif
  }
#if defined(HAVE_ASYNC_LDS)
  asm volatile("s_wait_asynccnt 0" ::: "memory");
#endif
  __syncthreads();
}

// ---------------- elementwise converts ----------------
__global__ void f32_to_f16_kernel(const float* __restrict__ in, _Float16* __restrict__ out, int n) {
  for (int i = blockIdx.x * blockDim.x + threadIdx.x; i < n; i += gridDim.x * blockDim.x)
    out[i] = (_Float16)in[i];
}

// in[K][N] (row-major) -> out[N][K] f16
__global__ void transpose_to_f16_kernel(const float* __restrict__ in, _Float16* __restrict__ out,
                                        int K, int N) {
  int total = K * N;
  for (int i = blockIdx.x * blockDim.x + threadIdx.x; i < total; i += gridDim.x * blockDim.x) {
    int n = i / K, k = i - n * K;
    out[i] = (_Float16)in[k * N + n];
  }
}

// ---------------- gates: softmax(x @ Wgate + bgate) over H ----------------
__global__ __launch_bounds__(128) void gates_kernel(const float* __restrict__ x,
                                                    const float* __restrict__ Wg,
                                                    const float* __restrict__ bg,
                                                    float* __restrict__ gates) {
  const int lane = threadIdx.x & 31;
  const int tok = blockIdx.x * (blockDim.x >> 5) + (threadIdx.x >> 5); // one wave per token
  const float* xp = x + (size_t)tok * C_DIM;
  float acc[H_DIM];
#pragma unroll
  for (int h = 0; h < H_DIM; ++h) acc[h] = 0.f;
  for (int k = lane; k < C_DIM; k += 32) {
    float xv = xp[k];
    const float* w = Wg + (size_t)k * H_DIM;
#pragma unroll
    for (int h = 0; h < H_DIM; ++h) acc[h] += xv * w[h];
  }
#pragma unroll
  for (int h = 0; h < H_DIM; ++h) {
#pragma unroll
    for (int off = 16; off > 0; off >>= 1) acc[h] += __shfl_xor(acc[h], off, 32);
    acc[h] += bg[h];
  }
  float m = -__builtin_inff();
#pragma unroll
  for (int h = 0; h < H_DIM; ++h) m = fmaxf(m, acc[h]);
  float s = 0.f;
  float e[H_DIM];
#pragma unroll
  for (int h = 0; h < H_DIM; ++h) { e[h] = __expf(acc[h] - m); s += e[h]; }
  float inv = 1.0f / s;
  if (lane == 0) {
#pragma unroll
    for (int h = 0; h < H_DIM; ++h) gates[(size_t)tok * H_DIM + h] = e[h] * inv;
  }
}

// Software-pipelined WMMA mainloop: A fragments from LDS tile, B fragments
// double-buffered from global so loads for k+1 overlap WMMAs for k.
__device__ __forceinline__ void gemm_mainloop(const _Float16* La,  // lane's LDS A row base
                                              const _Float16* bp,  // lane's global B base
                                              v8f& c0, v8f& c1, v8f& c2, v8f& c3) {
  v16h b0 = frag_ld(bp);
  v16h b1 = frag_ld(bp + 16 * C_DIM);
  v16h b2 = frag_ld(bp + 32 * C_DIM);
  v16h b3 = frag_ld(bp + 48 * C_DIM);
  for (int k0 = 32; k0 < C_DIM; k0 += 32) {
    v16h nb0 = frag_ld(bp + k0);
    v16h nb1 = frag_ld(bp + 16 * C_DIM + k0);
    v16h nb2 = frag_ld(bp + 32 * C_DIM + k0);
    v16h nb3 = frag_ld(bp + 48 * C_DIM + k0);
    v16h a = frag_ld(La + k0 - 32);
    c0 = WMMA_F16(a, b0, c0);
    c1 = WMMA_F16(a, b1, c1);
    c2 = WMMA_F16(a, b2, c2);
    c3 = WMMA_F16(a, b3, c3);
    b0 = nb0; b1 = nb1; b2 = nb2; b3 = nb3;
  }
  v16h a = frag_ld(La + C_DIM - 32);
  c0 = WMMA_F16(a, b0, c0);
  c1 = WMMA_F16(a, b1, c1);
  c2 = WMMA_F16(a, b2, c2);
  c3 = WMMA_F16(a, b3, c3);
}

// ---------------- GEMM1: qk = x @ Wqk + bqk, scatter to qh / kh / vT ----------------
__global__ __launch_bounds__(128) void gemm_qk_kernel(const _Float16* __restrict__ A,
                                                      const _Float16* __restrict__ Bt,
                                                      const float* __restrict__ bias,
                                                      _Float16* __restrict__ qh,
                                                      _Float16* __restrict__ kh,
                                                      _Float16* __restrict__ vT) {
  __shared__ __align__(16) _Float16 Atile[16 * C_DIM]; // 32 KB
  const int lane = threadIdx.x & 31;
  const int wid = blockIdx.x * 4 + (threadIdx.x >> 5);
  const int nGroups = (2 * C_DIM) / 64;   // 32 (divisible by 4 -> mTile uniform per block)
  const int mTile = wid / nGroups;
  const int nG = wid - mTile * nGroups;
  stage_a_tile(A + (size_t)mTile * 16 * C_DIM, Atile);
  const int nL = lane & 15;
  const int koff = (lane >> 4) * 8;
  const _Float16* La = Atile + nL * C_DIM + koff;
  const _Float16* bp = Bt + (size_t)(nG * 64 + nL) * C_DIM + koff;
  v8f c0 = {}, c1 = {}, c2 = {}, c3 = {};
  gemm_mainloop(La, bp, c0, c1, c2, c3);
  const int mOff = (lane >> 4) * 8;
  v8f cc[4] = {c0, c1, c2, c3};
#pragma unroll
  for (int nc = 0; nc < 4; ++nc) {
    const int col = nG * 64 + nc * 16 + nL;
    const float bv = bias[col];
#pragma unroll
    for (int v = 0; v < 8; ++v) {
      int r = mTile * 16 + v + mOff;
      float val = cc[nc][v] + bv;
      int b = r >> 10, t = r & (T_DIM - 1);
      if (col < C_DIM) { // q half
        int h = col >> 6, d = col & 63;
        qh[(((size_t)(b * H_DIM + h) * T_DIM) + t) * D_DIM + d] = (_Float16)val;
      } else {           // k half (v = k)
        int c2i = col - C_DIM;
        int h = c2i >> 6, d = c2i & 63;
        kh[(((size_t)(b * H_DIM + h) * T_DIM) + t) * D_DIM + d] = (_Float16)val;
        vT[((size_t)(b * H_DIM + h) * D_DIM + d) * T_DIM + t] = (_Float16)val;
      }
    }
  }
}

// ---------------- GEMM2: out = gated @ Wproj + bproj (f32 out) ----------------
__global__ __launch_bounds__(128) void gemm_proj_kernel(const _Float16* __restrict__ A,
                                                        const _Float16* __restrict__ Bt,
                                                        const float* __restrict__ bias,
                                                        float* __restrict__ out) {
  __shared__ __align__(16) _Float16 Atile[16 * C_DIM]; // 32 KB
  const int lane = threadIdx.x & 31;
  const int wid = blockIdx.x * 4 + (threadIdx.x >> 5);
  const int nGroups = C_DIM / 64;        // 16
  const int mTile = wid / nGroups;
  const int nG = wid - mTile * nGroups;
  stage_a_tile(A + (size_t)mTile * 16 * C_DIM, Atile);
  const int nL = lane & 15;
  const int koff = (lane >> 4) * 8;
  const _Float16* La = Atile + nL * C_DIM + koff;
  const _Float16* bp = Bt + (size_t)(nG * 64 + nL) * C_DIM + koff;
  v8f c0 = {}, c1 = {}, c2 = {}, c3 = {};
  gemm_mainloop(La, bp, c0, c1, c2, c3);
  const int mOff = (lane >> 4) * 8;
  v8f cc[4] = {c0, c1, c2, c3};
#pragma unroll
  for (int nc = 0; nc < 4; ++nc) {
    const int col = nG * 64 + nc * 16 + nL;
    const float bv = bias[col];
#pragma unroll
    for (int v = 0; v < 8; ++v) {
      int r = mTile * 16 + v + mOff;
      out[(size_t)r * C_DIM + col] = cc[nc][v] + bv;
    }
  }
}

// ---------------- flash attention with per-head windows + gating ----------------
__global__ __launch_bounds__(128) void attn_kernel(const _Float16* __restrict__ qh,
                                                   const _Float16* __restrict__ kh,
                                                   const _Float16* __restrict__ vT,
                                                   const float* __restrict__ gates,
                                                   _Float16* __restrict__ gatedh) {
  __shared__ __align__(16) _Float16 smem[4][16 * 32]; // per-wave P staging tile
  const int lane = threadIdx.x & 31;
  const int w = threadIdx.x >> 5;
  const int task = blockIdx.x * 4 + w;       // B*H*(T/16) = 4096 tasks
  const int qt = task & 63;
  const int bh = task >> 6;
  const int h = bh & (H_DIM - 1);
  const int b = bh >> 4;
  int Wh = T_DIM >> h; if (Wh < 4) Wh = 4;   // window for this head
  const int t0 = qt * 16;
  const int nL = lane & 15;
  const int koff = (lane >> 4) * 8;
  const int mOff = koff;                     // C-layout row offset is the same 0/8 split

  const _Float16* qb = qh + ((size_t)bh * T_DIM + t0) * D_DIM;
  const _Float16* kb = kh + (size_t)bh * T_DIM * D_DIM;
  const _Float16* vb = vT + (size_t)bh * D_DIM * T_DIM;

  // q fragments for the whole tile (D=64 -> two K=32 fragments)
  v16h a0 = frag_ld(qb + nL * D_DIM + 0 + koff);
  v16h a1 = frag_ld(qb + nL * D_DIM + 32 + koff);

  v8f o0 = {}, o1 = {}, o2 = {}, o3 = {};
  float mrow[8], lrow[8];
#pragma unroll
  for (int v = 0; v < 8; ++v) { mrow[v] = -__builtin_inff(); lrow[v] = 0.f; }

  int jStart = t0 - Wh + 1; if (jStart < 0) jStart = 0;
  jStart &= ~31;
  const float scale = 0.125f; // 1/sqrt(64)

  for (int j0 = jStart; j0 <= t0 + 15; j0 += 32) {
    // ---- S = q k^T for keys [j0, j0+31] ----
    v16h bk0lo = frag_ld(kb + (size_t)(j0 + nL) * D_DIM + koff);
    v16h bk0hi = frag_ld(kb + (size_t)(j0 + nL) * D_DIM + 32 + koff);
    v16h bk1lo = frag_ld(kb + (size_t)(j0 + 16 + nL) * D_DIM + koff);
    v16h bk1hi = frag_ld(kb + (size_t)(j0 + 16 + nL) * D_DIM + 32 + koff);
    v8f s0 = {}; s0 = WMMA_F16(a0, bk0lo, s0); s0 = WMMA_F16(a1, bk0hi, s0);
    v8f s1 = {}; s1 = WMMA_F16(a0, bk1lo, s1); s1 = WMMA_F16(a1, bk1hi, s1);

    _Float16* L = smem[w];
#pragma unroll
    for (int v = 0; v < 8; ++v) {
      const int i = t0 + v + mOff;
      const int ja = j0 + nL;
      const int jb = j0 + 16 + nL;
      const bool va = (ja <= i) && (ja > i - Wh);
      const bool vbm = (jb <= i) && (jb > i - Wh);
      float sa = va ? s0[v] * scale : -__builtin_inff();
      float sb = vbm ? s1[v] * scale : -__builtin_inff();
      // row max across the 16-lane group
      float t = fmaxf(sa, sb);
#pragma unroll
      for (int off = 8; off > 0; off >>= 1) t = fmaxf(t, __shfl_xor(t, off, 32));
      float mN = fmaxf(mrow[v], t);
      float alpha = (mrow[v] == mN) ? 1.0f : __expf(mrow[v] - mN);
      float pa = va ? __expf(sa - mN) : 0.f;
      float pb = vbm ? __expf(sb - mN) : 0.f;
      float rs = pa + pb;
#pragma unroll
      for (int off = 8; off > 0; off >>= 1) rs += __shfl_xor(rs, off, 32);
      lrow[v] = lrow[v] * alpha + rs;
      mrow[v] = mN;
      o0[v] *= alpha; o1[v] *= alpha; o2[v] *= alpha; o3[v] *= alpha;
      // stage P (C-layout -> row-major 16x32 f16 tile in LDS)
      const int r = v + mOff;
      L[r * 32 + nL] = (_Float16)pa;
      L[r * 32 + 16 + nL] = (_Float16)pb;
    }
    asm volatile("s_wait_dscnt 0" ::: "memory");
    // reload P as an A fragment (transpose across lanes via LDS)
    v16h pA = frag_ld(L + nL * 32 + koff);
    // ---- O += P V ; V rows contiguous thanks to vT[B,H,D,T] layout ----
    const _Float16* vp = vb + j0 + koff;
    o0 = WMMA_F16(pA, frag_ld(vp + (size_t)(0 + nL) * T_DIM), o0);
    o1 = WMMA_F16(pA, frag_ld(vp + (size_t)(16 + nL) * T_DIM), o1);
    o2 = WMMA_F16(pA, frag_ld(vp + (size_t)(32 + nL) * T_DIM), o2);
    o3 = WMMA_F16(pA, frag_ld(vp + (size_t)(48 + nL) * T_DIM), o3);
    asm volatile("s_wait_dscnt 0" ::: "memory");
  }

  // ---- epilogue: normalize, gate, store f16 for projection GEMM ----
#pragma unroll
  for (int v = 0; v < 8; ++v) {
    const int i = t0 + v + mOff;
    const float g = gates[((size_t)b * T_DIM + i) * H_DIM + h];
    const float inv = g / lrow[v];
    _Float16* op = gatedh + ((size_t)b * T_DIM + i) * C_DIM + h * D_DIM + nL;
    op[0]  = (_Float16)(o0[v] * inv);
    op[16] = (_Float16)(o1[v] * inv);
    op[32] = (_Float16)(o2[v] * inv);
    op[48] = (_Float16)(o3[v] * inv);
  }
}

extern "C" void kernel_launch(void* const* d_in, const int* in_sizes, int n_in,
                              void* d_out, int out_size, void* d_ws, size_t ws_size,
                              hipStream_t stream) {
  const float* x     = (const float*)d_in[0];
  const float* Wqk   = (const float*)d_in[1];
  const float* bqk   = (const float*)d_in[2];
  const float* Wgate = (const float*)d_in[3];
  const float* bgate = (const float*)d_in[4];
  const float* Wproj = (const float*)d_in[5];
  const float* bproj = (const float*)d_in[6];

  char* ws = (char*)d_ws;
  size_t off = 0;
  auto alloc = [&](size_t bytes) -> void* {
    void* p = ws + off;
    off += (bytes + 255) & ~(size_t)255;
    return p;
  };
  _Float16* xh     = (_Float16*)alloc((size_t)M_ROWS * C_DIM * 2);     // x in f16
  _Float16* wqkT   = (_Float16*)alloc((size_t)2 * C_DIM * C_DIM * 2);  // Wqk^T f16 [2C,C]
  _Float16* wprojT = (_Float16*)alloc((size_t)C_DIM * C_DIM * 2);      // Wproj^T f16 [C,C]
  _Float16* qh     = (_Float16*)alloc((size_t)B_DIM * H_DIM * T_DIM * D_DIM * 2);
  _Float16* kh     = (_Float16*)alloc((size_t)B_DIM * H_DIM * T_DIM * D_DIM * 2);
  _Float16* vT     = (_Float16*)alloc((size_t)B_DIM * H_DIM * D_DIM * T_DIM * 2);
  float*    gates  = (float*)alloc((size_t)B_DIM * T_DIM * H_DIM * 4);
  _Float16* gatedh = xh; // xh is dead after GEMM1; reuse for the gated output

  f32_to_f16_kernel<<<1024, 256, 0, stream>>>(x, xh, M_ROWS * C_DIM);
  transpose_to_f16_kernel<<<1024, 256, 0, stream>>>(Wqk, wqkT, C_DIM, 2 * C_DIM);
  transpose_to_f16_kernel<<<1024, 256, 0, stream>>>(Wproj, wprojT, C_DIM, C_DIM);
  gates_kernel<<<M_ROWS / 4, 128, 0, stream>>>(x, Wgate, bgate, gates);
  // waves = (4096/16) * (2048/64) = 8192 -> 2048 blocks of 4 waves
  gemm_qk_kernel<<<2048, 128, 0, stream>>>(xh, wqkT, bqk, qh, kh, vT);
  // tasks = B*H*(T/16) = 4096 -> 1024 blocks of 4 waves
  attn_kernel<<<1024, 128, 0, stream>>>(qh, kh, vT, gates, gatedh);
  // waves = (4096/16) * (1024/64) = 4096 -> 1024 blocks of 4 waves
  gemm_proj_kernel<<<1024, 128, 0, stream>>>(gatedh, wprojT, bproj, (float*)d_out);
}
